// LorentzNet_8366596292935
// MI455X (gfx1250) — compile-verified, matrix operand order checked by reading, not run on hardware
//
#include <hip/hip_runtime.h>
#include <math.h>

// ---------------------------------------------------------------------------
// LorentzNet forward, MI455X / gfx1250 (wave32, WMMA f16 -> f32 accumulate,
// TDM tensor_load_to_lds for weight-tile staging).
//
// Sizes (fixed by the reference):
#define BATCH   32
#define NP      64           // particles
#define EDGES   (NP*(NP-1))  // 4032, ordered i-major: edges [i*63, i*63+63) share edge_i==i
#define NSC     8
#define H       256
#define NLAYERS 6
#define BE      (BATCH*EDGES)   // 129024 edge rows
#define BN_ROWS (BATCH*NP)      // 2048 node rows
#define KE  514                 // 2H+2 (edge MLP input)
#define KH  520                 // 2H+NSC (node MLP input)
#define KP  520                 // padded K (multiple of 8 halves -> 16B rows)

typedef __attribute__((ext_vector_type(16))) _Float16 v16h;
typedef __attribute__((ext_vector_type(8)))  _Float16 v8h;
typedef __attribute__((ext_vector_type(8)))  float    v8f;

#ifndef USE_TDM
#  if defined(__has_builtin)
#    if __has_builtin(__builtin_amdgcn_tensor_load_to_lds)
#      define USE_TDM 1
#    else
#      define USE_TDM 0
#    endif
#  else
#    define USE_TDM 0
#  endif
#endif

// ---------------------------------------------------------------------------
// CDNA5 16-bit WMMA fragment layout (ISA 7.12.2):
//  A 16x32 f16:  lane 0-15 -> M=lane,    K in {0..7} U {16..23}
//                lane16-31 -> M=lane-16, K in {8..15} U {24..31}
//  B 32x16 f16:  same (lane,element)->K mapping, N = lane&15 (K-major in LDS)
//  C/D 16x16 f32: lane 0-15 -> N=lane, M=vgpr g; lane16-31 -> N=lane-16, M=g+8
// rowp points at the 32-half K-run for this row; kb = (lane&16)?8:0.
__device__ __forceinline__ v16h load_frag(const _Float16* rowp, int kb) {
  v8h lo = *(const v8h*)(rowp + kb);        // K = kb .. kb+7
  v8h hi = *(const v8h*)(rowp + kb + 16);   // K = kb+16 .. kb+23
  v16h r;
#pragma unroll
  for (int i = 0; i < 8; ++i) { r[i] = lo[i]; r[i + 8] = hi[i]; }
  return r;
}

// Fallback vector staging of one 256(N) x 32(K) tile of an N-major f16 weight
// [n][Kp] into sB[256][40].
__device__ __forceinline__ void stage_B32(_Float16* sB, const _Float16* Wt,
                                          int Kp, int k0, int tid) {
  for (int i = tid; i < 256 * 4; i += 128) {
    int n = i >> 2, seg = (i & 3) << 3, k = k0 + seg;
    v8h v;
    if (k + 7 < Kp) {
      v = *(const v8h*)(Wt + (size_t)n * Kp + k);
    } else {
#pragma unroll
      for (int t = 0; t < 8; ++t) v[t] = (k + t < Kp) ? Wt[(size_t)n * Kp + k + t] : (_Float16)0;
    }
    *(v8h*)(sB + n * 40 + seg) = v;
  }
}

#if USE_TDM
typedef __attribute__((ext_vector_type(4))) unsigned int u32x4;
typedef __attribute__((ext_vector_type(4))) int          i32x4;
typedef __attribute__((ext_vector_type(8))) int          i32x8;

// TDM: DMA one 32(K-halves) x 256(N-rows) weight tile into sB[256][40].
// D# group0: count=1, lds_addr, 57-bit global tile address, type=2.
// D# group1: data_size=2B; pad_enable with interval 16 DWORDs (=64B row) and
//            amount 4 DWORDs (=16B) -> reproduces the 40-half LDS row stride;
//            tensor_dim0 = Kp-k0 so the K-tail reads return zero (TDM OOB);
//            tile 32 x 256, dim0 stride = Kp elements.
__device__ __forceinline__ void tdm_load_B(const _Float16* Wt, unsigned lds_byte_off,
                                           int Kp, int k0) {
  unsigned long long ga = (unsigned long long)(uintptr_t)Wt + (unsigned long long)k0 * 2ull;
  unsigned dim0 = (unsigned)(Kp - k0);
  u32x4 g0;
  g0[0] = 1u;                                                   // count=1 (user D#)
  g0[1] = lds_byte_off;                                         // lds_addr
  g0[2] = (unsigned)ga;                                         // global_addr[31:0]
  g0[3] = (unsigned)((ga >> 32) & 0x01ffffffull) | (2u << 30);  // addr[56:32] | type=2
  i32x8 g1;
  g1[0] = (int)((1u << 16)      // data_size = 2 bytes
              | (1u << 20)      // pad_enable
              | (3u << 22)      // pad_interval: 16 DWORDs (one 64B tile row)
              | (3u << 25));    // pad_amount:   4 DWORDs (16B)
  g1[1] = (int)((dim0 & 0xffffu) << 16);                        // tensor_dim0 lo
  g1[2] = (int)(((dim0 >> 16) & 0xffffu) | (256u << 16));       // dim0 hi | tensor_dim1 lo
  g1[3] = (int)(32u << 16);                                     // dim1 hi=0 | tile_dim0=32
  g1[4] = (int)256;                                             // tile_dim1=256, tile_dim2=0
  g1[5] = (int)(unsigned)Kp;                                    // tensor_dim0_stride lo
  g1[6] = 0;                                                    // stride hi | dim1_stride lo
  g1[7] = 0;
  i32x4 z4 = {0, 0, 0, 0};
#if __clang_major__ >= 23
  i32x8 z8 = {0, 0, 0, 0, 0, 0, 0, 0};
  __builtin_amdgcn_tensor_load_to_lds(g0, g1, z4, z4, z8, 0);
#else
  __builtin_amdgcn_tensor_load_to_lds(g0, g1, z4, z4, 0);
#endif
}
#define TDM_ISSUE(Wt, sBp, Kp, k0) do { if (wave == 0) tdm_load_B((Wt), (unsigned)(uintptr_t)(sBp), (Kp), (k0)); } while (0)
#define TDM_WAIT(Wt, sBp, Kp, k0)  do { if (wave == 0) __builtin_amdgcn_s_wait_tensorcnt(0); } while (0)
#else
#define TDM_ISSUE(Wt, sBp, Kp, k0) do { } while (0)
#define TDM_WAIT(Wt, sBp, Kp, k0)  stage_B32((sBp), (Wt), (Kp), (k0), tid)
#endif

__device__ __forceinline__ float psi_f(float t) { return copysignf(log1pf(fabsf(t)), t); }

// ---------------------------------------------------------------------------
// h = scalars @ emb_W + emb_b  (K=8 -> plain VALU), also f16 shadow copy.
__global__ void k_embed(const float* __restrict__ sc, const float* __restrict__ W,
                        const float* __restrict__ b, float* __restrict__ h,
                        _Float16* __restrict__ hf) {
  int idx = blockIdx.x * 256 + threadIdx.x;     // over BN_ROWS*H
  int r = idx >> 8, c = idx & 255;
  float v = b[c];
#pragma unroll
  for (int k = 0; k < NSC; ++k) v += sc[r * NSC + k] * W[k * H + c];
  h[idx] = v; hf[idx] = (_Float16)v;
}

// Convert+transpose fp32 weight [K][N=256] -> f16 [n][Kp], zero-padded K..Kp.
__global__ void k_wt(_Float16* __restrict__ dst, const float* __restrict__ src,
                     int K, int Kp) {
  int idx = blockIdx.x * 256 + threadIdx.x;
  if (idx >= 256 * Kp) return;
  int n = idx / Kp, k = idx - n * Kp;
  dst[idx] = (k < K) ? (_Float16)src[(size_t)k * 256 + n] : (_Float16)0;
}

// Per-edge Minkowski features + x_diff.
__global__ void k_edge_geom(const float* __restrict__ x, const int* __restrict__ ei,
                            const int* __restrict__ ej, float* __restrict__ norms,
                            float* __restrict__ dots, float* __restrict__ xdiff) {
  int idx = blockIdx.x * 256 + threadIdx.x;
  if (idx >= BE) return;
  int b = idx / EDGES, e = idx - b * EDGES;
  const float* xi = x + ((size_t)b * NP + ei[e]) * 4;
  const float* xj = x + ((size_t)b * NP + ej[e]) * 4;
  float d0 = xi[0] - xj[0], d1 = xi[1] - xj[1], d2 = xi[2] - xj[2], d3 = xi[3] - xj[3];
  float n2 = d0 * d0 - d1 * d1 - d2 * d2 - d3 * d3;                 // 2*d0^2 - sum d^2
  float dt = xi[0] * xj[0] - xi[1] * xj[1] - xi[2] * xj[2] - xi[3] * xj[3];
  norms[idx] = psi_f(n2); dots[idx] = psi_f(dt);
  xdiff[idx * 4 + 0] = d0; xdiff[idx * 4 + 1] = d1;
  xdiff[idx * 4 + 2] = d2; xdiff[idx * 4 + 3] = d3;
}

// ---------------------------------------------------------------------------
// GEMM1 (edge): z1 = [h_i | h_j | norms | dots] @ We1, with BN sum/sumsq atomics.
// 128 threads (4 waves) -> 32 edge rows x 256 cols; K = 514 pad 544 (17x32).
// Per wave/chunk: 2 A-frags + 4 B-frags -> 8 WMMAs.
__global__ void k_gemm_e1(const _Float16* __restrict__ hf, const int* __restrict__ ei,
                          const int* __restrict__ ej, const float* __restrict__ norms,
                          const float* __restrict__ dots, const _Float16* __restrict__ W1t,
                          float* __restrict__ z1, float* __restrict__ gsum,
                          float* __restrict__ gsq) {
  __shared__ _Float16 sA[32 * 40];
  __shared__ _Float16 sB[256 * 40];
  __shared__ int sEi[32], sEj[32];
  __shared__ _Float16 sNrm[32], sDot[32];
  const int tid = threadIdx.x, lane = tid & 31, wave = tid >> 5;
  const int m = lane & 15, kb = (lane & 16) ? 8 : 0;
  const int r0 = blockIdx.x * 32;          // EDGES%32==0 -> block never crosses batch
  const int bb = r0 / EDGES;
  if (tid < 32) {
    int ee = (r0 - bb * EDGES) + tid;
    sEi[tid] = ei[ee]; sEj[tid] = ej[ee];
    sNrm[tid] = (_Float16)norms[r0 + tid]; sDot[tid] = (_Float16)dots[r0 + tid];
  }
  __syncthreads();
  const _Float16* hb = hf + (size_t)bb * NP * H;
  v8f acc[2][4];
#pragma unroll
  for (int rf = 0; rf < 2; ++rf)
#pragma unroll
    for (int f = 0; f < 4; ++f) acc[rf][f] = (v8f){0, 0, 0, 0, 0, 0, 0, 0};

  for (int kc = 0; kc < 17; ++kc) {
    const int k0 = kc * 32;
    TDM_ISSUE(W1t, sB, KP, k0);
    if (k0 < 512) {                       // gather h_i (k<256) or h_j (256<=k<512)
      const bool hi = (k0 >= 256);
      for (int i = tid; i < 32 * 4; i += 128) {
        int row = i >> 2, seg = (i & 3) << 3;
        int node = hi ? sEj[row] : sEi[row];
        v8h v = *(const v8h*)(hb + (size_t)node * H + (k0 - (hi ? 256 : 0)) + seg);
        *(v8h*)(sA + row * 40 + seg) = v;
      }
    } else {                              // tail: norms, dots, zero pad
      for (int i = tid; i < 32 * 32; i += 128) {
        int row = i >> 5, kk = i & 31, k = k0 + kk;
        _Float16 v = (_Float16)0;
        if (k == 512) v = sNrm[row]; else if (k == 513) v = sDot[row];
        sA[row * 40 + kk] = v;
      }
    }
    TDM_WAIT(W1t, sB, KP, k0);
    __syncthreads();
    v16h a0 = load_frag(sA + m * 40, kb);
    v16h a1 = load_frag(sA + (16 + m) * 40, kb);
#pragma unroll
    for (int f = 0; f < 4; ++f) {
      v16h bf = load_frag(sB + (wave * 64 + f * 16 + m) * 40, kb);
      acc[0][f] = __builtin_amdgcn_wmma_f32_16x16x32_f16(false, a0, false, bf,
                                                         (short)0, acc[0][f], false, false);
      acc[1][f] = __builtin_amdgcn_wmma_f32_16x16x32_f16(false, a1, false, bf,
                                                         (short)0, acc[1][f], false, false);
    }
    __syncthreads();
  }
#pragma unroll
  for (int rf = 0; rf < 2; ++rf)
#pragma unroll
    for (int f = 0; f < 4; ++f) {
      int col = wave * 64 + f * 16 + m;
      float s = 0.f, q = 0.f;
#pragma unroll
      for (int g = 0; g < 8; ++g) {
        float v = acc[rf][f][g];
        z1[(size_t)(r0 + rf * 16 + kb + g) * H + col] = v;
        s += v; q += v * v;
      }
      atomicAdd(&gsum[col], s);
      atomicAdd(&gsq[col], q);
    }
}

// Fold BN stats with gamma/beta -> per-channel (a,b): bn(z) = a*z + b.
__global__ void k_bn_fin(const float* __restrict__ sum, const float* __restrict__ sq,
                         const float* __restrict__ g, const float* __restrict__ be,
                         float inv_count, float* __restrict__ a, float* __restrict__ bo) {
  int c = threadIdx.x;
  float mu = sum[c] * inv_count;
  float var = sq[c] * inv_count - mu * mu;
  float ac = g[c] * rsqrtf(var + 1e-5f);
  a[c] = ac; bo[c] = be[c] - mu * ac;
}

// ---------------------------------------------------------------------------
// Edge MLP tail: m1=relu(bn(z1)); m2=relu(m1@We2+be2); m=m2*sigmoid(m2@Wm+bm);
// if do_x: phi=relu(m@Wx1+bx1)@Wx2; trans=clip(x_diff*phi,+-100).
__global__ void k_edge_mlp(const float* __restrict__ z1, const float* __restrict__ ea,
                           const float* __restrict__ ebv, const _Float16* __restrict__ W2t,
                           const float* __restrict__ be2, const float* __restrict__ Wm,
                           const float* __restrict__ bm, const _Float16* __restrict__ Wx1t,
                           const float* __restrict__ bx1, const float* __restrict__ Wx2,
                           const float* __restrict__ xdiff, _Float16* __restrict__ mbuf,
                           float* __restrict__ transb, int do_x) {
  __shared__ _Float16 sA[16 * 264];
  __shared__ _Float16 sB[256 * 40];
  __shared__ float sM[16 * 264];
  __shared__ float sRed[128];
  __shared__ float sGate[16];
  const int tid = threadIdx.x, lane = tid & 31, wave = tid >> 5;
  const int m = lane & 15, kb = (lane & 16) ? 8 : 0;
  const int r0 = blockIdx.x * 16;

  for (int i = tid; i < 16 * 256; i += 128) {          // A = relu(bn(z1)) in f16
    int row = i >> 8, c = i & 255;
    float v = z1[(size_t)(r0 + row) * H + c];
    v = fmaxf(v * ea[c] + ebv[c], 0.f);
    sA[row * 264 + c] = (_Float16)v;
  }
  __syncthreads();
  v8f acc[4];
#pragma unroll
  for (int f = 0; f < 4; ++f) acc[f] = (v8f){0, 0, 0, 0, 0, 0, 0, 0};
  for (int kc = 0; kc < 8; ++kc) {                     // m2 = A @ We2
    const int k0 = kc * 32;
    TDM_ISSUE(W2t, sB, 256, k0);
    TDM_WAIT(W2t, sB, 256, k0);
    __syncthreads();
    v16h a = load_frag(sA + m * 264 + k0, kb);
#pragma unroll
    for (int f = 0; f < 4; ++f) {
      v16h bf = load_frag(sB + (wave * 64 + f * 16 + m) * 40, kb);
      acc[f] = __builtin_amdgcn_wmma_f32_16x16x32_f16(false, a, false, bf,
                                                      (short)0, acc[f], false, false);
    }
    __syncthreads();
  }
#pragma unroll
  for (int f = 0; f < 4; ++f) {
    int col = wave * 64 + f * 16 + m;
#pragma unroll
    for (int g = 0; g < 8; ++g)
      sM[(kb + g) * 264 + col] = fmaxf(acc[f][g] + be2[col], 0.f);
  }
  __syncthreads();
  {                                                    // gate = sigmoid(m2 . Wm + bm)
    int row = tid >> 3, seg = (tid & 7) * 32;
    float p = 0.f;
    for (int c = seg; c < seg + 32; ++c) p += sM[row * 264 + c] * Wm[c];
    sRed[tid] = p;
  }
  __syncthreads();
  if (tid < 16) {
    float g = bm[0];
#pragma unroll
    for (int t = 0; t < 8; ++t) g += sRed[tid * 8 + t];
    sGate[tid] = 1.f / (1.f + expf(-g));
  }
  __syncthreads();
  for (int i = tid; i < 16 * 256; i += 128) {          // m -> mbuf + next A
    int row = i >> 8, c = i & 255;
    float mv = sM[row * 264 + c] * sGate[row];
    mbuf[(size_t)(r0 + row) * H + c] = (_Float16)mv;
    sA[row * 264 + c] = (_Float16)mv;
  }
  __syncthreads();
  if (do_x) {                                          // phi_x branch
#pragma unroll
    for (int f = 0; f < 4; ++f) acc[f] = (v8f){0, 0, 0, 0, 0, 0, 0, 0};
    for (int kc = 0; kc < 8; ++kc) {
      const int k0 = kc * 32;
      TDM_ISSUE(Wx1t, sB, 256, k0);
      TDM_WAIT(Wx1t, sB, 256, k0);
      __syncthreads();
      v16h a = load_frag(sA + m * 264 + k0, kb);
#pragma unroll
      for (int f = 0; f < 4; ++f) {
        v16h bf = load_frag(sB + (wave * 64 + f * 16 + m) * 40, kb);
        acc[f] = __builtin_amdgcn_wmma_f32_16x16x32_f16(false, a, false, bf,
                                                        (short)0, acc[f], false, false);
      }
      __syncthreads();
    }
#pragma unroll
    for (int f = 0; f < 4; ++f) {
      int col = wave * 64 + f * 16 + m;
#pragma unroll
      for (int g = 0; g < 8; ++g)
        sM[(kb + g) * 264 + col] = fmaxf(acc[f][g] + bx1[col], 0.f);
    }
    __syncthreads();
    {
      int row = tid >> 3, seg = (tid & 7) * 32;
      float p = 0.f;
      for (int c = seg; c < seg + 32; ++c) p += sM[row * 264 + c] * Wx2[c];
      sRed[tid] = p;
    }
    __syncthreads();
    if (tid < 16) {
      float ph = 0.f;
#pragma unroll
      for (int t = 0; t < 8; ++t) ph += sRed[tid * 8 + t];
      sGate[tid] = ph;
    }
    __syncthreads();
    if (tid < 64) {
      int row = tid >> 2, d = tid & 3;
      float tr = xdiff[(size_t)(r0 + row) * 4 + d] * sGate[row];
      transb[(size_t)(r0 + row) * 4 + d] = fminf(fmaxf(tr, -100.f), 100.f);
    }
  }
}

// Segmented reductions over the 63 contiguous edges with edge_i == i.
__global__ void k_agg_h(const _Float16* __restrict__ mbuf, float* __restrict__ aggh) {
  int idx = blockIdx.x * 256 + threadIdx.x;            // over BN_ROWS*H
  int b = idx / (NP * H), rem = idx - b * NP * H, i = rem >> 8, c = rem & 255;
  size_t base = ((size_t)b * EDGES + (size_t)i * (NP - 1)) * H + c;
  float s = 0.f;
  for (int t = 0; t < NP - 1; ++t) s += (float)mbuf[base + (size_t)t * H];
  aggh[idx] = s;
}

__global__ void k_x_update(const float* __restrict__ transb, float* __restrict__ x) {
  int idx = blockIdx.x * 256 + threadIdx.x;            // over BN_ROWS*4
  if (idx >= BN_ROWS * 4) return;
  int b = idx / (NP * 4), rem = idx - b * NP * 4, i = rem >> 2, d = rem & 3;
  size_t base = ((size_t)b * EDGES + (size_t)i * (NP - 1)) * 4 + d;
  float s = 0.f;
  for (int t = 0; t < NP - 1; ++t) s += transb[base + (size_t)t * 4];
  x[idx] += s * (0.001f / (float)(NP - 1));            // C_WEIGHT / (n-1)
}

// ---------------------------------------------------------------------------
// Node GEMM1: zh = [h | agg_h | scalars] @ Wh1 + bh1, with BN-stats atomics.
// 32 node rows per workgroup.
__global__ void k_gemm_h1(const _Float16* __restrict__ hf, const float* __restrict__ aggh,
                          const float* __restrict__ sc, const _Float16* __restrict__ W1t,
                          const float* __restrict__ bh1, float* __restrict__ zh,
                          float* __restrict__ nsum, float* __restrict__ nsq) {
  __shared__ _Float16 sA[32 * 40];
  __shared__ _Float16 sB[256 * 40];
  const int tid = threadIdx.x, lane = tid & 31, wave = tid >> 5;
  const int m = lane & 15, kb = (lane & 16) ? 8 : 0;
  const int r0 = blockIdx.x * 32;
  v8f acc[2][4];
#pragma unroll
  for (int rf = 0; rf < 2; ++rf)
#pragma unroll
    for (int f = 0; f < 4; ++f) acc[rf][f] = (v8f){0, 0, 0, 0, 0, 0, 0, 0};
  for (int kc = 0; kc < 17; ++kc) {
    const int k0 = kc * 32;
    TDM_ISSUE(W1t, sB, KP, k0);
    for (int i = tid; i < 32 * 32; i += 128) {
      int row = i >> 5, kk = i & 31, k = k0 + kk, r = r0 + row;
      _Float16 v;
      if (k < 256)       v = hf[(size_t)r * H + k];
      else if (k < 512)  v = (_Float16)aggh[(size_t)r * H + (k - 256)];
      else if (k < KH)   v = (_Float16)sc[(size_t)r * NSC + (k - 512)];
      else               v = (_Float16)0;
      sA[row * 40 + kk] = v;
    }
    TDM_WAIT(W1t, sB, KP, k0);
    __syncthreads();
    v16h a0 = load_frag(sA + m * 40, kb);
    v16h a1 = load_frag(sA + (16 + m) * 40, kb);
#pragma unroll
    for (int f = 0; f < 4; ++f) {
      v16h bf = load_frag(sB + (wave * 64 + f * 16 + m) * 40, kb);
      acc[0][f] = __builtin_amdgcn_wmma_f32_16x16x32_f16(false, a0, false, bf,
                                                         (short)0, acc[0][f], false, false);
      acc[1][f] = __builtin_amdgcn_wmma_f32_16x16x32_f16(false, a1, false, bf,
                                                         (short)0, acc[1][f], false, false);
    }
    __syncthreads();
  }
#pragma unroll
  for (int rf = 0; rf < 2; ++rf)
#pragma unroll
    for (int f = 0; f < 4; ++f) {
      int col = wave * 64 + f * 16 + m;
      float s = 0.f, q = 0.f;
#pragma unroll
      for (int g = 0; g < 8; ++g) {
        float v = acc[rf][f][g] + bh1[col];
        zh[(size_t)(r0 + rf * 16 + kb + g) * H + col] = v;
        s += v; q += v * v;
      }
      atomicAdd(&nsum[col], s);
      atomicAdd(&nsq[col], q);
    }
}

// Node GEMM2 (residual): h += relu(bn(zh)) @ Wh2 + bh2; refresh f16 shadow.
__global__ void k_gemm_h2(const float* __restrict__ zh, const float* __restrict__ ha,
                          const float* __restrict__ hbv, const _Float16* __restrict__ W2t,
                          const float* __restrict__ bh2, float* __restrict__ h,
                          _Float16* __restrict__ hf) {
  __shared__ _Float16 sA[16 * 264];
  __shared__ _Float16 sB[256 * 40];
  const int tid = threadIdx.x, lane = tid & 31, wave = tid >> 5;
  const int m = lane & 15, kb = (lane & 16) ? 8 : 0;
  const int r0 = blockIdx.x * 16;
  for (int i = tid; i < 16 * 256; i += 128) {
    int row = i >> 8, c = i & 255;
    float v = fmaxf(zh[(size_t)(r0 + row) * H + c] * ha[c] + hbv[c], 0.f);
    sA[row * 264 + c] = (_Float16)v;
  }
  __syncthreads();
  v8f acc[4];
#pragma unroll
  for (int f = 0; f < 4; ++f) acc[f] = (v8f){0, 0, 0, 0, 0, 0, 0, 0};
  for (int kc = 0; kc < 8; ++kc) {
    const int k0 = kc * 32;
    TDM_ISSUE(W2t, sB, 256, k0);
    TDM_WAIT(W2t, sB, 256, k0);
    __syncthreads();
    v16h a = load_frag(sA + m * 264 + k0, kb);
#pragma unroll
    for (int f = 0; f < 4; ++f) {
      v16h bf = load_frag(sB + (wave * 64 + f * 16 + m) * 40, kb);
      acc[f] = __builtin_amdgcn_wmma_f32_16x16x32_f16(false, a, false, bf,
                                                      (short)0, acc[f], false, false);
    }
    __syncthreads();
  }
#pragma unroll
  for (int f = 0; f < 4; ++f) {
    int col = wave * 64 + f * 16 + m;
#pragma unroll
    for (int g = 0; g < 8; ++g) {
      size_t o = (size_t)(r0 + kb + g) * H + col;
      float v = h[o] + acc[f][g] + bh2[col];
      h[o] = v; hf[o] = (_Float16)v;
    }
  }
}

// Head: mean over particles, relu(dec1), dec2. One block per batch element.
__global__ void k_head(const float* __restrict__ h, const float* __restrict__ W1,
                       const float* __restrict__ b1, const float* __restrict__ W2,
                       const float* __restrict__ b2, float* __restrict__ out) {
  __shared__ float sRow[H];
  __shared__ float sT[H];
  int b = blockIdx.x, c = threadIdx.x;
  float s = 0.f;
  for (int n = 0; n < NP; ++n) s += h[((size_t)b * NP + n) * H + c];
  sRow[c] = s * (1.f / NP);
  __syncthreads();
  float v = b1[c];
  for (int k = 0; k < H; ++k) v += sRow[k] * W1[(size_t)k * H + c];
  sT[c] = fmaxf(v, 0.f);
  __syncthreads();
  if (c < 2) {
    float o = b2[c];
    for (int k = 0; k < H; ++k) o += sT[k] * W2[k * 2 + c];
    out[b * 2 + c] = o;
  }
}

// ---------------------------------------------------------------------------
// Input flattening assumption: top-level dict insertion order
//   (scalars, x, <params leaves>, edge_i, edge_j)
// with params flattened as a jax pytree (keys sorted at every level).
enum {
  IN_SCALARS = 0, IN_X = 1,
  IN_DEC_W1 = 2, IN_DEC_W2 = 3, IN_DEC_B1 = 4, IN_DEC_B2 = 5,
  IN_EMB_W = 6, IN_EMB_B = 7,
  IN_L0 = 8,                     // 16 sorted leaves per layer
  IN_EDGE_I = IN_L0 + NLAYERS * 16,   // 104
  IN_EDGE_J = IN_EDGE_I + 1           // 105
};
enum { LW_We1 = 0, LW_We2, LW_Wh1, LW_Wh2, LW_Wm, LW_Wx1, LW_Wx2,
       LW_be2, LW_bh1, LW_bh2, LW_bm,
       LW_bn_e_b, LW_bn_e_g, LW_bn_h_b, LW_bn_h_g, LW_bx1 };

extern "C" void kernel_launch(void* const* d_in, const int* in_sizes, int n_in,
                              void* d_out, int out_size, void* d_ws, size_t ws_size,
                              hipStream_t stream) {
  (void)in_sizes; (void)n_in; (void)out_size; (void)ws_size;
  const float* scalars = (const float*)d_in[IN_SCALARS];
  const float* x_in    = (const float*)d_in[IN_X];
  const int*   edi     = (const int*)d_in[IN_EDGE_I];
  const int*   edj     = (const int*)d_in[IN_EDGE_J];
  auto L = [&](int l, int k) -> const float* {
    return (const float*)d_in[IN_L0 + l * 16 + k];
  };

  // -------- workspace bump allocator --------
  size_t off = 0;
  auto take = [&](size_t bytes) -> void* {
    void* p = (char*)d_ws + off;
    off = (off + bytes + 255) & ~(size_t)255;
    return p;
  };
  float*    xbuf  = (float*)take((size_t)BN_ROWS * 4 * 4);
  float*    h     = (float*)take((size_t)BN_ROWS * H * 4);
  _Float16* hfb   = (_Float16*)take((size_t)BN_ROWS * H * 2);
  float*    aggh  = (float*)take((size_t)BN_ROWS * H * 4);
  float*    zh    = (float*)take((size_t)BN_ROWS * H * 4);
  float*    norms = (float*)take((size_t)BE * 4);
  float*    dots  = (float*)take((size_t)BE * 4);
  float*    xdiff = (float*)take((size_t)BE * 4 * 4);
  float*    transb= (float*)take((size_t)BE * 4 * 4);
  float*    z1    = (float*)take((size_t)BE * H * 4);       // 132 MB
  _Float16* mbuf  = (_Float16*)take((size_t)BE * H * 2);    // 66 MB
  float*    stats = (float*)take(8 * 256 * 4);
  // stats layout: [0]=gsum [256]=gsq [512]=ea [768]=eb [1024]=nsum [1280]=nsq [1536]=ha [1792]=hb
  _Float16* We1t  = (_Float16*)take((size_t)256 * KP * 2);
  _Float16* We2t  = (_Float16*)take((size_t)256 * 256 * 2);
  _Float16* Wx1t  = (_Float16*)take((size_t)256 * 256 * 2);
  _Float16* Wh1t  = (_Float16*)take((size_t)256 * KP * 2);
  _Float16* Wh2t  = (_Float16*)take((size_t)256 * 256 * 2);

  (void)hipMemcpyAsync(xbuf, x_in, (size_t)BN_ROWS * 4 * 4,
                       hipMemcpyDeviceToDevice, stream);
  k_embed<<<BN_ROWS, 256, 0, stream>>>(scalars, (const float*)d_in[IN_EMB_W],
                                       (const float*)d_in[IN_EMB_B], h, hfb);

  const int gWT520 = (256 * KP + 255) / 256;
  const int gWT256 = 256;
  for (int l = 0; l < NLAYERS; ++l) {
    const int do_x = (l != NLAYERS - 1);
    // f16 N-major weights (hit L2 on every GEMM workgroup / TDM tile load)
    k_wt<<<gWT520, 256, 0, stream>>>(We1t, L(l, LW_We1), KE, KP);
    k_wt<<<gWT256, 256, 0, stream>>>(We2t, L(l, LW_We2), 256, 256);
    if (do_x) k_wt<<<gWT256, 256, 0, stream>>>(Wx1t, L(l, LW_Wx1), 256, 256);
    k_wt<<<gWT520, 256, 0, stream>>>(Wh1t, L(l, LW_Wh1), KH, KP);
    k_wt<<<gWT256, 256, 0, stream>>>(Wh2t, L(l, LW_Wh2), 256, 256);

    k_edge_geom<<<(BE + 255) / 256, 256, 0, stream>>>(xbuf, edi, edj, norms, dots, xdiff);

    (void)hipMemsetAsync(stats, 0, 512 * 4, stream);
    k_gemm_e1<<<BE / 32, 128, 0, stream>>>(hfb, edi, edj, norms, dots, We1t,
                                           z1, stats, stats + 256);
    k_bn_fin<<<1, 256, 0, stream>>>(stats, stats + 256, L(l, LW_bn_e_g), L(l, LW_bn_e_b),
                                    1.f / (float)BE, stats + 512, stats + 768);
    k_edge_mlp<<<BE / 16, 128, 0, stream>>>(z1, stats + 512, stats + 768, We2t,
                                            L(l, LW_be2), L(l, LW_Wm), L(l, LW_bm),
                                            Wx1t, L(l, LW_bx1), L(l, LW_Wx2),
                                            xdiff, mbuf, transb, do_x);
    k_agg_h<<<BN_ROWS, 256, 0, stream>>>(mbuf, aggh);
    if (do_x) k_x_update<<<(BN_ROWS * 4 + 255) / 256, 256, 0, stream>>>(transb, xbuf);

    (void)hipMemsetAsync(stats + 1024, 0, 512 * 4, stream);
    k_gemm_h1<<<BN_ROWS / 32, 128, 0, stream>>>(hfb, aggh, scalars, Wh1t, L(l, LW_bh1),
                                                zh, stats + 1024, stats + 1280);
    k_bn_fin<<<1, 256, 0, stream>>>(stats + 1024, stats + 1280, L(l, LW_bn_h_g),
                                    L(l, LW_bn_h_b), 1.f / (float)BN_ROWS,
                                    stats + 1536, stats + 1792);
    k_gemm_h2<<<BN_ROWS / 16, 128, 0, stream>>>(zh, stats + 1536, stats + 1792, Wh2t,
                                                L(l, LW_bh2), h, hfb);
  }

  k_head<<<BATCH, 256, 0, stream>>>(h, (const float*)d_in[IN_DEC_W1],
                                    (const float*)d_in[IN_DEC_B1],
                                    (const float*)d_in[IN_DEC_W2],
                                    (const float*)d_in[IN_DEC_B2], (float*)d_out);
}
